// SWTBlock_58042188038805
// MI455X (gfx1250) — matrix operand order repaired and checked by reading, not compiled, not run on hardware
//
#include <hip/hip_runtime.h>
#include <math.h>

#define C_DIM 384
#define HID_DIM 1536
#define HEADS 12
#define DHEAD 32
#define WINL 64
#define IMGHW 128
#define LDA 392   // padded row stride (ushorts) for 384-wide LDS tiles
#define LDP 72    // padded row stride for 64-wide P tiles

typedef __attribute__((ext_vector_type(16))) __bf16 v16bf;
typedef __attribute__((ext_vector_type(8))) float v8f;
typedef __attribute__((ext_vector_type(4))) unsigned int ui32x4;
typedef __attribute__((ext_vector_type(8))) int i32x8;
typedef __attribute__((ext_vector_type(4))) int i32x4;

union BF16x16 { v16bf v; unsigned short u[16]; };

__device__ __forceinline__ unsigned short f2bf(float f) {
  unsigned int u = __float_as_uint(f);
  u += 0x7FFFu + ((u >> 16) & 1u);            // round-to-nearest-even
  return (unsigned short)(u >> 16);
}

__device__ __forceinline__ v8f vzero() {
  v8f z;
#pragma unroll
  for (int i = 0; i < 8; ++i) z[i] = 0.0f;
  return z;
}

__device__ __forceinline__ v8f wmma_bf16(v16bf a, v16bf b, v8f c) {
  return __builtin_amdgcn_wmma_f32_16x16x32_bf16(false, a, false, b, (short)0, c,
                                                 false, false);
}

// A fragment (16x32 bf16), storage row-major [rows][ld].
__device__ __forceinline__ v16bf load_A_frag(const unsigned short* base, int ld,
                                             int row0, int k0, int lane) {
  int row = row0 + (lane & 15);
  int half = lane >> 4;
  const unsigned short* p = base + row * ld + k0 + half * 16;
  BF16x16 f;
#pragma unroll
  for (int v = 0; v < 8; ++v) {
    int kp = 2 * v;
    f.u[2 * v]     = p[kp];
    f.u[2 * v + 1] = p[kp + 1];
  }
  return f.v;
}

// B fragment (32x16 bf16) from [K][N] row-major storage.
__device__ __forceinline__ v16bf load_B_lds(const unsigned short* base, int ld,
                                            int k0, int col0, int lane) {
  int col = col0 + (lane & 15);
  int half = lane >> 4;
  const unsigned short* p = base + (k0 + half * 16) * ld + col;
  BF16x16 f;
#pragma unroll
  for (int i = 0; i < 16; ++i) f.u[i] = p[i * ld];
  return f.v;
}

// B fragment where storage is [N][K] row-major (B = storage^T): contiguous per lane.
__device__ __forceinline__ v16bf load_BT_lds(const unsigned short* base, int ld,
                                             int k0, int col0, int lane) {
  int col = col0 + (lane & 15);
  int half = lane >> 4;
  const unsigned short* p = base + col * ld + k0 + half * 16;
  BF16x16 f;
#pragma unroll
  for (int i = 0; i < 16; ++i) f.u[i] = p[i];
  return f.v;
}

// B fragment from pre-packed weights: one 32-byte contiguous load per lane.
__device__ __forceinline__ v16bf load_B_packed(const unsigned short* __restrict__ base,
                                               int NT, int kt, int nt, int lane) {
  const unsigned short* p = base + (((size_t)(kt * NT + nt) * 32 + lane) << 4);
  BF16x16 f;
#pragma unroll
  for (int i = 0; i < 16; ++i) f.u[i] = p[i];
  return f.v;
}

// ---------------- GroupNorm(1,C) statistics: two-pass deterministic ----------------

__global__ void gn_partial_kernel(const float* __restrict__ src, float* __restrict__ part) {
  const int n = blockIdx.y, cid = blockIdx.x, tid = threadIdx.x;
  const size_t per = (size_t)C_DIM * IMGHW * IMGHW;   // 6291456
  const size_t len = per >> 6;                        // 64 chunks
  const size_t s0 = (size_t)n * per + (size_t)cid * len;
  float s = 0.f, q = 0.f;
  for (size_t i = tid; i < len; i += 256) {
    float v = src[s0 + i];
    s += v; q += v * v;
  }
  __shared__ float ss[256], sq[256];
  ss[tid] = s; sq[tid] = q;
  __syncthreads();
  for (int off = 128; off > 0; off >>= 1) {
    if (tid < off) { ss[tid] += ss[tid + off]; sq[tid] += sq[tid + off]; }
    __syncthreads();
  }
  if (tid == 0) {
    part[(n * 64 + cid) * 2 + 0] = ss[0];
    part[(n * 64 + cid) * 2 + 1] = sq[0];
  }
}

__global__ void gn_final_kernel(const float* __restrict__ part, float* __restrict__ mr) {
  const int n = blockIdx.x, t = threadIdx.x;  // 8 blocks x 64 threads
  __shared__ float ss[64], sq[64];
  ss[t] = part[(n * 64 + t) * 2 + 0];
  sq[t] = part[(n * 64 + t) * 2 + 1];
  __syncthreads();
  for (int off = 32; off > 0; off >>= 1) {
    if (t < off) { ss[t] += ss[t + off]; sq[t] += sq[t + off]; }
    __syncthreads();
  }
  if (t == 0) {
    const float inv = 1.0f / (float)((size_t)C_DIM * IMGHW * IMGHW);
    float mean = ss[0] * inv;
    float var = sq[0] * inv - mean * mean;
    mr[2 * n] = mean;
    mr[2 * n + 1] = rsqrtf(var + 1e-5f);
  }
}

// ------------- Pack weight [N][K] fp32 -> bf16 B-fragment-linear layout -------------

__global__ void pack_b_kernel(const float* __restrict__ w, unsigned short* __restrict__ dst,
                              int K, int N) {
  int idx = blockIdx.x * 256 + threadIdx.x;
  if (idx >= K * N) return;
  int i = idx & 15;
  int lane = (idx >> 4) & 31;
  int fr = idx >> 9;                 // frag id = kt*NT + nt
  int NT = N >> 4;
  int kt = fr / NT, nt = fr - kt * NT;
  int halfb = lane >> 4, nlb = lane & 15;
  int k = kt * 32 + halfb * 16 + i;  // f.u[i] <-> k = half*16+i
  int col = nt * 16 + nlb;
  dst[idx] = f2bf(w[(size_t)col * K + k]);
}

// ---------------- Fused shifted-window attention block (one window/block) ----------------

__global__ __launch_bounds__(256) void attn_kernel(
    const float* __restrict__ x, const float* __restrict__ mr1,
    const float* __restrict__ n1w, const float* __restrict__ n1b,
    const unsigned short* __restrict__ qkvt, const float* __restrict__ qkvb,
    const unsigned short* __restrict__ projt, const float* __restrict__ projb,
    float* __restrict__ x2) {
  __shared__ unsigned short Qs[WINL * LDA];
  __shared__ unsigned short Ks[WINL * LDA];
  __shared__ unsigned short Vs[WINL * LDA];
  __shared__ unsigned short As[WINL * LDA];    // A (norm. input), later O
  __shared__ unsigned short Ps[8][WINL * LDP]; // per-wave softmax tile

  const int tid = threadIdx.x;
  const int lane = tid & 31;
  const int wv = tid >> 5;
  const int nl = lane & 15;
  const int half = lane >> 4;

  const int wid = blockIdx.x;          // 2048 windows
  const int n = wid >> 8;
  const int wh = (wid >> 4) & 15;
  const int ww = wid & 15;

  const float mean = mr1[2 * n], rstd = mr1[2 * n + 1];

  // Phase 0: normalize + shift(-4,-4) + window-partition -> As (bf16 [64][384])
  for (int idx = tid; idx < WINL * C_DIM; idx += 256) {
    int c = idx >> 6, l = idx & 63;
    int hr = ((wh << 3) + (l >> 3) + 4) & 127;
    int wr = ((ww << 3) + (l & 7) + 4) & 127;
    float v = x[(((size_t)n * C_DIM + c) << 14) + (hr << 7) + wr];
    v = (v - mean) * rstd * n1w[c] + n1b[c];
    As[l * LDA + c] = f2bf(v);
  }
  __syncthreads();

  // Phase 1: QKV = A @ Wqkv^T : 72 n-tiles, 9 per wave, K = 12 steps of 32
  for (int j = 0; j < 9; ++j) {
    int ntile = wv * 9 + j;
    v8f acc[4];
    for (int m = 0; m < 4; ++m) acc[m] = vzero();
    for (int k = 0; k < 12; ++k) {
      v16bf bf = load_B_packed(qkvt, 72, k, ntile, lane);
#pragma unroll
      for (int m = 0; m < 4; ++m) {
        v16bf af = load_A_frag(As, LDA, m * 16, k * 32, lane);
        acc[m] = wmma_bf16(af, bf, acc[m]);
      }
    }
    int o = ntile * 16 + nl;
    float bia = qkvb[o];
    unsigned short* dst = (o < 384) ? Qs : ((o < 768) ? Ks : Vs);
    int ch = (o < 384) ? o : ((o < 768) ? o - 384 : o - 768);
#pragma unroll
    for (int m = 0; m < 4; ++m)
#pragma unroll
      for (int r = 0; r < 8; ++r) {
        int row = m * 16 + half * 8 + r;
        dst[row * LDA + ch] = f2bf(acc[m][r] + bia);
      }
  }
  __syncthreads();

  // Phase 2: per-head attention (waves 0..7 take heads wv, wv+8)
  const float scale = 0.17677669529663687f;  // 1/sqrt(32)
  unsigned short* Pb = Ps[wv];
  for (int h = wv; h < HEADS; h += 8) {
    for (int mt = 0; mt < 4; ++mt) {
      v8f s[4];
      for (int nt = 0; nt < 4; ++nt) s[nt] = vzero();
      v16bf qf = load_A_frag(Qs, LDA, mt * 16, h * 32, lane);
      for (int nt = 0; nt < 4; ++nt) {
        v16bf kf = load_BT_lds(Ks, LDA, h * 32, nt * 16, lane);
        s[nt] = wmma_bf16(qf, kf, s[nt]);
      }
#pragma unroll
      for (int r = 0; r < 8; ++r) {
        float mx = -3.0e38f;
        for (int nt = 0; nt < 4; ++nt) mx = fmaxf(mx, s[nt][r]);
        for (int off = 1; off < 16; off <<= 1) mx = fmaxf(mx, __shfl_xor(mx, off, 32));
        float e[4], sum = 0.0f;
        for (int nt = 0; nt < 4; ++nt) {
          e[nt] = __expf((s[nt][r] - mx) * scale);
          sum += e[nt];
        }
        for (int off = 1; off < 16; off <<= 1) sum += __shfl_xor(sum, off, 32);
        float inv = 1.0f / sum;
        int row = mt * 16 + half * 8 + r;
        for (int nt = 0; nt < 4; ++nt) Pb[row * LDP + nt * 16 + nl] = f2bf(e[nt] * inv);
      }
    }
    for (int mt = 0; mt < 4; ++mt)
      for (int nt = 0; nt < 2; ++nt) {
        v8f o = vzero();
        for (int kk = 0; kk < 2; ++kk) {
          v16bf pf = load_A_frag(Pb, LDP, mt * 16, kk * 32, lane);
          v16bf vf = load_B_lds(Vs, LDA, kk * 32, h * 32 + nt * 16, lane);
          o = wmma_bf16(pf, vf, o);
        }
#pragma unroll
        for (int r = 0; r < 8; ++r) {
          int row = mt * 16 + half * 8 + r;
          As[row * LDA + h * 32 + nt * 16 + nl] = f2bf(o[r]);
        }
      }
  }
  __syncthreads();

  // Phase 3: proj + window-reverse + unshift + residual -> x2
  for (int j = 0; j < 3; ++j) {
    int ntile = wv * 3 + j;  // 24 n-tiles of 16
    v8f acc[4];
    for (int m = 0; m < 4; ++m) acc[m] = vzero();
    for (int k = 0; k < 12; ++k) {
      v16bf bf = load_B_packed(projt, 24, k, ntile, lane);
#pragma unroll
      for (int m = 0; m < 4; ++m) {
        v16bf af = load_A_frag(As, LDA, m * 16, k * 32, lane);
        acc[m] = wmma_bf16(af, bf, acc[m]);
      }
    }
    int o = ntile * 16 + nl;
    float bia = projb[o];
#pragma unroll
    for (int m = 0; m < 4; ++m)
#pragma unroll
      for (int r = 0; r < 8; ++r) {
        int l = m * 16 + half * 8 + r;
        int hr = ((wh << 3) + (l >> 3) + 4) & 127;
        int wr = ((ww << 3) + (l & 7) + 4) & 127;
        size_t gi = (((size_t)n * C_DIM + o) << 14) + (hr << 7) + wr;
        x2[gi] = x[gi] + acc[m][r] + bia;
      }
  }
}

// ---------------- Fused MLP: norm2 -> fc1 -> GELU -> fc2 -> +x2, in place ----------------

__global__ __launch_bounds__(256) void mlp_kernel(
    float* __restrict__ xio, const float* __restrict__ mr2,
    const float* __restrict__ n2w, const float* __restrict__ n2b,
    const unsigned short* __restrict__ fc1t, const float* __restrict__ fc1b,
    const unsigned short* __restrict__ fc2t, const float* __restrict__ fc2b) {
  __shared__ float Xraw[WINL * C_DIM];       // raw x2 tile: [c][l], 98304 B
  __shared__ unsigned short Ab[WINL * LDA];  // normalized bf16 [l][c]
  __shared__ unsigned short Hb[WINL * LDA];  // one 384-wide hidden chunk

  const int tid = threadIdx.x;
  const int lane = tid & 31;
  const int wv = tid >> 5;
  const int nl = lane & 15;
  const int half = lane >> 4;

  const int n = blockIdx.x >> 8;             // 256 blocks of 64 tokens per image
  const int pix0 = (blockIdx.x & 255) << 6;
  const float mean = mr2[2 * n], rstd = mr2[2 * n + 1];

  // Phase 0a: TDM async copy global -> LDS (wave 0 issues; 2D tile 64 x 384 fp32,
  // row stride 16384 elements). D# layout per CDNA5 ISA 8.3/8.4.
  if (wv == 0) {
    unsigned long long ga =
        (unsigned long long)(const void*)(xio + (((size_t)n * C_DIM) << 14) + pix0);
    unsigned lds_off = (unsigned)(unsigned long long)(const void*)&Xraw[0];
    ui32x4 g0;
    g0[0] = 1u;                                    // count=1, user desc, no gather
    g0[1] = lds_off;                               // lds_addr
    g0[2] = (unsigned)ga;                          // global_addr[31:0]
    g0[3] = (unsigned)((ga >> 32) & 0x1FFFFFFu)    // global_addr[56:32]
            | (2u << 30);                          // type=2 ("image")
    i32x8 g1;
    g1[0] = 0x00020000;        // workgroup_mask=0, data_size=2 (4B), no flags
    g1[1] = (int)(64u << 16);  // atomic_barrier_addr=0 | tensor_dim0[15:0]=64
    g1[2] = (int)(384u << 16); // tensor_dim0[31:16]=0 | tensor_dim1[15:0]=384
    g1[3] = (int)(64u << 16);  // tensor_dim1[31:16]=0 | tile_dim0=64
    g1[4] = 384;               // tile_dim1=384, tile_dim2=0
    g1[5] = 16384;             // tensor_dim0_stride[31:0] = 16384 elements
    g1[6] = 0;                 // dim0_stride[47:32]=0, dim1_stride[15:0]=0
    g1[7] = 0;                 // dim1_stride[47:16]=0
    i32x4 gz;
    gz[0] = 0; gz[1] = 0; gz[2] = 0; gz[3] = 0;    // groups 2/3 unused (2D)
    i32x8 gz8;
#pragma unroll
    for (int i = 0; i < 8; ++i) gz8[i] = 0;        // extra group (6-arg form)
    __builtin_amdgcn_tensor_load_to_lds(g0, g1, gz, gz, gz8, 0);
    __builtin_amdgcn_s_wait_tensorcnt(0);
  }
  __syncthreads();

  // Phase 0b: normalize LDS fp32 -> bf16 A tile (transposed to [l][c])
  for (int idx = tid; idx < WINL * C_DIM; idx += 256) {
    int c = idx >> 6, l = idx & 63;
    float v = Xraw[c * 64 + l];
    v = (v - mean) * rstd * n2w[c] + n2b[c];
    Ab[l * LDA + c] = f2bf(v);
  }
  __syncthreads();

  v8f oacc[3][4];
  for (int j = 0; j < 3; ++j)
    for (int m = 0; m < 4; ++m) oacc[j][m] = vzero();

  for (int cc = 0; cc < 4; ++cc) {  // 4 chunks of 384 hidden units
    if (cc) __syncthreads();        // previous fc2 reads of Hb done
    for (int j = 0; j < 3; ++j) {
      int ntloc = wv * 3 + j;
      v8f acc[4];
      for (int m = 0; m < 4; ++m) acc[m] = vzero();
      for (int k = 0; k < 12; ++k) {
        v16bf bf = load_B_packed(fc1t, 96, k, cc * 24 + ntloc, lane);
#pragma unroll
        for (int m = 0; m < 4; ++m) {
          v16bf af = load_A_frag(Ab, LDA, m * 16, k * 32, lane);
          acc[m] = wmma_bf16(af, bf, acc[m]);
        }
      }
      float bia = fc1b[cc * 384 + ntloc * 16 + nl];
#pragma unroll
      for (int m = 0; m < 4; ++m)
#pragma unroll
        for (int r = 0; r < 8; ++r) {
          float v = acc[m][r] + bia;
          v = 0.5f * v * (1.0f + erff(v * 0.70710678118654752f));  // exact GELU
          int row = m * 16 + half * 8 + r;
          Hb[row * LDA + ntloc * 16 + nl] = f2bf(v);
        }
    }
    __syncthreads();
    for (int j = 0; j < 3; ++j) {
      int ntile = wv * 3 + j;
      for (int k = 0; k < 12; ++k) {
        v16bf bf = load_B_packed(fc2t, 24, cc * 12 + k, ntile, lane);
#pragma unroll
        for (int m = 0; m < 4; ++m) {
          v16bf af = load_A_frag(Hb, LDA, m * 16, k * 32, lane);
          oacc[j][m] = wmma_bf16(af, bf, oacc[j][m]);
        }
      }
    }
  }

  // Epilogue: out = x2 (from LDS tile) + fc2 + bias
  for (int j = 0; j < 3; ++j) {
    int o = (wv * 3 + j) * 16 + nl;
    float bia = fc2b[o];
#pragma unroll
    for (int m = 0; m < 4; ++m)
#pragma unroll
      for (int r = 0; r < 8; ++r) {
        int l = m * 16 + half * 8 + r;
        size_t gi = (((size_t)n * C_DIM + o) << 14) + pix0 + l;
        xio[gi] = Xraw[o * 64 + l] + oacc[j][m][r] + bia;
      }
  }
}

// ---------------------------------- launch ----------------------------------

extern "C" void kernel_launch(void* const* d_in, const int* in_sizes, int n_in,
                              void* d_out, int out_size, void* d_ws, size_t ws_size,
                              hipStream_t stream) {
  (void)in_sizes; (void)n_in; (void)out_size; (void)ws_size;
  const float* x     = (const float*)d_in[0];
  const float* n1w   = (const float*)d_in[1];
  const float* n1b   = (const float*)d_in[2];
  const float* qkvw  = (const float*)d_in[3];
  const float* qkvb  = (const float*)d_in[4];
  const float* projw = (const float*)d_in[5];
  const float* projb = (const float*)d_in[6];
  const float* n2w   = (const float*)d_in[7];
  const float* n2b   = (const float*)d_in[8];
  const float* fc1w  = (const float*)d_in[9];
  const float* fc1b  = (const float*)d_in[10];
  const float* fc2w  = (const float*)d_in[11];
  const float* fc2b  = (const float*)d_in[12];
  float* out = (float*)d_out;  // used as x2 scratch, then final output in place

  char* ws = (char*)d_ws;
  float* part = (float*)(ws + 0);                           // 4096 B
  float* mr1  = (float*)(ws + 4096);                        // 64 B
  float* mr2  = (float*)(ws + 4352);                        // 64 B
  unsigned short* qkvt  = (unsigned short*)(ws + 4608);     // 384x1152 bf16
  unsigned short* projt = (unsigned short*)(ws + 889344);   // 384x384
  unsigned short* fc1t  = (unsigned short*)(ws + 1184256);  // 384x1536
  unsigned short* fc2t  = (unsigned short*)(ws + 2363904);  // 1536x384

  gn_partial_kernel<<<dim3(64, 8), 256, 0, stream>>>(x, part);
  gn_final_kernel<<<8, 64, 0, stream>>>(part, mr1);

  pack_b_kernel<<<(384 * 1152 + 255) / 256, 256, 0, stream>>>(qkvw, qkvt, 384, 1152);
  pack_b_kernel<<<(384 * 384 + 255) / 256, 256, 0, stream>>>(projw, projt, 384, 384);
  pack_b_kernel<<<(384 * 1536 + 255) / 256, 256, 0, stream>>>(fc1w, fc1t, 384, 1536);
  pack_b_kernel<<<(1536 * 384 + 255) / 256, 256, 0, stream>>>(fc2w, fc2t, 1536, 384);

  attn_kernel<<<2048, 256, 0, stream>>>(x, mr1, n1w, n1b, qkvt, qkvb, projt, projb, out);

  gn_partial_kernel<<<dim3(64, 8), 256, 0, stream>>>(out, part);
  gn_final_kernel<<<8, 64, 0, stream>>>(part, mr2);

  mlp_kernel<<<2048, 256, 0, stream>>>(out, mr2, n2w, n2b, fc1t, fc1b, fc2t, fc2b);
}
